// CCNet_unit_68685116997898
// MI455X (gfx1250) — matrix operand order repaired and checked by reading, not compiled, not run on hardware
//
#include <hip/hip_runtime.h>
#include <hip/hip_bf16.h>

// ---------------------------------------------------------------------------
// CCNet block on gfx1250 (MI455X): all contractions via v_wmma_f32_16x16x32_bf16
// NT-GEMM formulation so both WMMA fragments are contiguous 16B loads.
// Round 3: each wave owns TWO 16-row M strips (block = 128 M x 64 N), so every
// 4 B-fragments feed 8 WMMAs -> half the redundant B traffic per FLOP and more
// independent WMMAs to fill co-exec hazard slots. K-loop stays double-buffered.
// ---------------------------------------------------------------------------

typedef __attribute__((ext_vector_type(16))) __bf16 v16bf;
typedef __attribute__((ext_vector_type(8)))  float  v8f;

#define NEGBIG (-1000000000.0f)

union FragBF { uint4 u4[2]; v16bf v; };

__device__ __forceinline__ v8f wmma_bf16(v16bf a, v16bf b, v8f c) {
  // (neg_a, A, neg_b, B, c_mod, C, reuse_a, reuse_b)
  return __builtin_amdgcn_wmma_f32_16x16x32_bf16(false, a, false, b, (short)0, c,
                                                 false, false);
}

// A fragment: lane row = lane&15; two 16B runs at +kbase and +kbase+16.
__device__ __forceinline__ void load_a(FragBF& f, const __bf16* arow, int kb,
                                       bool valid) {
  if (valid) {
    f.u4[0] = *(const uint4*)(arow + (kb << 5));
    f.u4[1] = *(const uint4*)(arow + (kb << 5) + 16);
  } else {
    f.u4[0] = make_uint4(0u, 0u, 0u, 0u);
    f.u4[1] = make_uint4(0u, 0u, 0u, 0u);
  }
}

// B fragment: two 16B runs at +kbase and +kbase+16 of its N-row.
__device__ __forceinline__ void load_b(FragBF& f, const __bf16* bp) {
  f.u4[0] = *(const uint4*)(bp);
  f.u4[1] = *(const uint4*)(bp + 16);
}

// ============================ data prep kernels =============================

// x: NCHW f32 (4,2048,64,64) -> xbf NHWC bf16 [b][h][w][ci]
__global__ __launch_bounds__(256) void cvt_x_kernel(const float* __restrict__ x,
                                                    __bf16* __restrict__ xbf) {
  size_t idx = (size_t)blockIdx.x * 256 + threadIdx.x;   // 4*2048*4096 = 2^25
  if (idx >= (size_t)4 * 2048 * 4096) return;
  int hw = (int)(idx & 4095);
  int ci = (int)((idx >> 12) & 2047);
  int b  = (int)(idx >> 23);
  xbf[((size_t)((b << 12) + hw)) * 2048 + ci] = (__bf16)x[idx];
}

// conv weight OIHW f32 -> [co][tap*Cin + ci] bf16 (N-major rows of length 9*Cin)
__global__ __launch_bounds__(256) void prep_w3_kernel(const float* __restrict__ w,
                                                      __bf16* __restrict__ wt,
                                                      int Cin, int total) {
  size_t idx = (size_t)blockIdx.x * 256 + threadIdx.x;   // over Cout*Cin*9
  if (idx >= (size_t)total) return;
  int t = (int)(idx % 9);
  size_t rest = idx / 9;
  int ci = (int)(rest % Cin);
  size_t co = rest / Cin;
  wt[co * (size_t)(9 * Cin) + (size_t)t * Cin + ci] = (__bf16)w[idx];
}

// plain f32 -> bf16 (1x1 weights are already [N][K] row-major)
__global__ __launch_bounds__(256) void cvt_bf_kernel(const float* __restrict__ in,
                                                     __bf16* __restrict__ out, int n) {
  int idx = blockIdx.x * 256 + threadIdx.x;
  if (idx < n) out[idx] = (__bf16)in[idx];
}

// v NHWC -> vtH[b][w][c][i=h], vtW[b][h][c][j=w]  (for NT aggregation GEMMs)
__global__ __launch_bounds__(256) void transpose_v_kernel(const __bf16* __restrict__ V,
                                                          __bf16* __restrict__ vtH,
                                                          __bf16* __restrict__ vtW) {
  size_t idx = (size_t)blockIdx.x * 256 + threadIdx.x;   // 4*64*64*512 = 2^23
  if (idx >= (size_t)4 * 64 * 64 * 512) return;
  int c = (int)(idx & 511);
  int w = (int)((idx >> 9) & 63);
  int h = (int)((idx >> 15) & 63);
  int b = (int)(idx >> 21);
  __bf16 val = V[idx];
  vtH[((size_t)((b << 6) + w) * 512 + c) * 64 + h] = val;
  vtW[((size_t)((b << 6) + h) * 512 + c) * 64 + w] = val;
}

// ====================== generic NT WMMA GEMM (conv / 1x1) ===================
// O[m,n] = sum_taps sum_k A_shift[m,k] * Bw[n, t*KC+k]
// Block = 128 M x 64 N; wave = two 16-row strips x 64 N (8 accumulators).
// MODE 0: +bias[n]          -> bf16 NHWC out
// MODE 1: BN+ReLU           -> bf16 NHWC out
// MODE 2: BN+ReLU           -> f32 NCHW out (final output)
template <int TAPS, int KC, int MODE>
__global__ __launch_bounds__(128) void gemm_nt_wmma(
    const __bf16* __restrict__ A, const __bf16* __restrict__ Bw,
    int Cout,
    const float* __restrict__ bias,
    const float* __restrict__ bng, const float* __restrict__ bnb,
    const float* __restrict__ bnm, const float* __restrict__ bnv,
    __bf16* __restrict__ outBF, float* __restrict__ outF32) {
  const int tid   = threadIdx.x;
  const int lane  = tid & 31;
  const int wid   = tid >> 5;               // 4 waves, 32 M-rows each
  const int mBase = blockIdx.x << 7;        // 128 M rows per block
  const int nBase = blockIdx.y << 6;
  constexpr int LDB = TAPS * KC;
  constexpr int NK  = KC >> 5;              // 32-wide K blocks per tap (even)

  const int khalf = lane >> 4;
  const int kbase = khalf << 3;

  int w0[2], h0[2], b0[2];
#pragma unroll
  for (int s = 0; s < 2; ++s) {
    const int mrow = mBase + (wid << 5) + (s << 4) + (lane & 15);
    w0[s] = mrow & 63;
    h0[s] = (mrow >> 6) & 63;
    b0[s] = mrow >> 12;
  }

  v8f acc[2][4];
  const v8f vz = {0.f, 0.f, 0.f, 0.f, 0.f, 0.f, 0.f, 0.f};
#pragma unroll
  for (int s = 0; s < 2; ++s)
#pragma unroll
    for (int j = 0; j < 4; ++j) acc[s][j] = vz;

  const __bf16* brow[4];
#pragma unroll
  for (int j = 0; j < 4; ++j)
    brow[j] = Bw + (size_t)(nBase + (j << 4) + (lane & 15)) * LDB + (khalf << 3);

#pragma unroll
  for (int t = 0; t < TAPS; ++t) {
    const int dh = (TAPS == 9) ? (t / 3 - 1) : 0;   // constant per unrolled t
    const int dw = (TAPS == 9) ? (t % 3 - 1) : 0;
    bool valid[2];
    const __bf16* arow[2];
#pragma unroll
    for (int s = 0; s < 2; ++s) {
      const int hh = h0[s] + dh, ww = w0[s] + dw;
      valid[s] = (hh >= 0) && (hh < 64) && (ww >= 0) && (ww < 64);
      arow[s] = A + ((size_t)(((b0[s] << 6) + hh) << 6) + ww) * KC + kbase;
    }
    const int koff = t * KC;

    // --- double-buffered K pipeline: load kb+1 while computing kb ---
    FragBF aE[2], aO[2], bE[4], bO[4];
#pragma unroll
    for (int s = 0; s < 2; ++s) load_a(aE[s], arow[s], 0, valid[s]);
#pragma unroll
    for (int j = 0; j < 4; ++j) load_b(bE[j], brow[j] + koff);

#pragma unroll 1
    for (int kb = 0; kb < NK; kb += 2) {
      // stage odd block
#pragma unroll
      for (int s = 0; s < 2; ++s) load_a(aO[s], arow[s], kb + 1, valid[s]);
#pragma unroll
      for (int j = 0; j < 4; ++j)
        load_b(bO[j], brow[j] + koff + ((kb + 1) << 5));
      // compute even block: 8 independent WMMAs
#pragma unroll
      for (int s = 0; s < 2; ++s)
#pragma unroll
        for (int j = 0; j < 4; ++j)
          acc[s][j] = wmma_bf16(aE[s].v, bE[j].v, acc[s][j]);
      // stage next even block
      if (kb + 2 < NK) {
#pragma unroll
        for (int s = 0; s < 2; ++s) load_a(aE[s], arow[s], kb + 2, valid[s]);
#pragma unroll
        for (int j = 0; j < 4; ++j)
          load_b(bE[j], brow[j] + koff + ((kb + 2) << 5));
      }
      // compute odd block
#pragma unroll
      for (int s = 0; s < 2; ++s)
#pragma unroll
        for (int j = 0; j < 4; ++j)
          acc[s][j] = wmma_bf16(aO[s].v, bO[j].v, acc[s][j]);
    }
  }

#pragma unroll
  for (int j = 0; j < 4; ++j) {
    const int n = nBase + (j << 4) + (lane & 15);
    float sc, sh;
    if (MODE == 0) {
      sc = 1.0f; sh = bias[n];
    } else {
      const float s = bng[n] * rsqrtf(bnv[n] + 1e-5f);
      sc = s; sh = bnb[n] - bnm[n] * s;
    }
#pragma unroll
    for (int s = 0; s < 2; ++s) {
#pragma unroll
      for (int r = 0; r < 8; ++r) {
        const int m = mBase + (wid << 5) + (s << 4) + (khalf << 3) + r;
        float val = acc[s][j][r] * sc + sh;
        if (MODE != 0) val = fmaxf(val, 0.0f);
        if (MODE == 2) {
          const int wq = m & 63, hq = (m >> 6) & 63, bq = m >> 12;
          outF32[(((size_t)bq * Cout + n) << 12) + (hq << 6) + wq] = val;
        } else {
          outBF[(size_t)m * Cout + n] = (__bf16)val;
        }
      }
    }
  }
}

// ======================= criss-cross energy GEMMs ===========================
// Shared 64x64x64 NT-GEMM core: all 10 fragments loaded up front, then 8 WMMAs.
__device__ __forceinline__ void gemm64_core(const __bf16* arow,
                                            const __bf16* const brow[4],
                                            v8f acc[4]) {
  FragBF a0, a1, b00, b01, b02, b03, b10, b11, b12, b13;
  a0.u4[0] = *(const uint4*)(arow);
  a0.u4[1] = *(const uint4*)(arow + 16);
  a1.u4[0] = *(const uint4*)(arow + 32);
  a1.u4[1] = *(const uint4*)(arow + 48);
  load_b(b00, brow[0]);       load_b(b10, brow[0] + 32);
  load_b(b01, brow[1]);       load_b(b11, brow[1] + 32);
  load_b(b02, brow[2]);       load_b(b12, brow[2] + 32);
  load_b(b03, brow[3]);       load_b(b13, brow[3] + 32);
  acc[0] = wmma_bf16(a0.v, b00.v, acc[0]);
  acc[1] = wmma_bf16(a0.v, b01.v, acc[1]);
  acc[2] = wmma_bf16(a0.v, b02.v, acc[2]);
  acc[3] = wmma_bf16(a0.v, b03.v, acc[3]);
  acc[0] = wmma_bf16(a1.v, b10.v, acc[0]);
  acc[1] = wmma_bf16(a1.v, b11.v, acc[1]);
  acc[2] = wmma_bf16(a1.v, b12.v, acc[2]);
  acc[3] = wmma_bf16(a1.v, b13.v, acc[3]);
}

// eH[b,h,w,i] = sum_c q[b,h,w,c] k[b,i,w,c]   (one block per (b,w), 64x64x64)
__global__ __launch_bounds__(128) void energy_h_kernel(const __bf16* __restrict__ Q,
                                                       const __bf16* __restrict__ Kb,
                                                       float* __restrict__ attn) {
  const int tid = threadIdx.x, lane = tid & 31, wid = tid >> 5;
  const int b = blockIdx.x >> 6, w = blockIdx.x & 63;
  const int khalf = lane >> 4, kbase = khalf << 3;
  const int h = (wid << 4) + (lane & 15);

  const __bf16* arow = Q + (((size_t)((b << 6) + h) << 6) + w) * 64 + kbase;
  const __bf16* brow[4];
#pragma unroll
  for (int j = 0; j < 4; ++j) {
    const int i = (j << 4) + (lane & 15);
    brow[j] = Kb + (((size_t)((b << 6) + i) << 6) + w) * 64 + (khalf << 3);
  }

  v8f acc[4];
  const v8f vz = {0.f, 0.f, 0.f, 0.f, 0.f, 0.f, 0.f, 0.f};
#pragma unroll
  for (int j = 0; j < 4; ++j) acc[j] = vz;
  gemm64_core(arow, brow, acc);

#pragma unroll
  for (int j = 0; j < 4; ++j) {
    const int i = (j << 4) + (lane & 15);
#pragma unroll
    for (int r = 0; r < 8; ++r) {
      const int hq = (wid << 4) + (khalf << 3) + r;
      float val = acc[j][r];
      if (i == hq) val += NEGBIG;                       // diagonal mask
      attn[(((size_t)((b << 6) + hq) << 6) + w) * 128 + i] = val;
    }
  }
}

// eW[b,h,w,j] = sum_c q[b,h,w,c] k[b,h,j,c]   (one block per (b,h))
__global__ __launch_bounds__(128) void energy_w_kernel(const __bf16* __restrict__ Q,
                                                       const __bf16* __restrict__ Kb,
                                                       float* __restrict__ attn) {
  const int tid = threadIdx.x, lane = tid & 31, wid = tid >> 5;
  const int b = blockIdx.x >> 6, h = blockIdx.x & 63;
  const int khalf = lane >> 4, kbase = khalf << 3;
  const int w = (wid << 4) + (lane & 15);

  const __bf16* arow = Q + (((size_t)((b << 6) + h) << 6) + w) * 64 + kbase;
  const __bf16* brow[4];
#pragma unroll
  for (int j = 0; j < 4; ++j) {
    const int jc = (j << 4) + (lane & 15);
    brow[j] = Kb + (((size_t)((b << 6) + h) << 6) + jc) * 64 + (khalf << 3);
  }

  v8f acc[4];
  const v8f vz = {0.f, 0.f, 0.f, 0.f, 0.f, 0.f, 0.f, 0.f};
#pragma unroll
  for (int j = 0; j < 4; ++j) acc[j] = vz;
  gemm64_core(arow, brow, acc);

#pragma unroll
  for (int j = 0; j < 4; ++j) {
    const int jc = (j << 4) + (lane & 15);
#pragma unroll
    for (int r = 0; r < 8; ++r) {
      const int wq = (wid << 4) + (khalf << 3) + r;
      attn[(((size_t)((b << 6) + h) << 6) + wq) * 128 + 64 + jc] = acc[j][r];
    }
  }
}

// softmax over 128-way concat; one wave per (b,h,w) row; bf16 output
__global__ __launch_bounds__(128) void softmax_kernel(const float* __restrict__ attn,
                                                      __bf16* __restrict__ abf) {
  const int tid = threadIdx.x, lane = tid & 31, wid = tid >> 5;
  const size_t row = (size_t)blockIdx.x * 4 + wid;      // < 16384
  const float* p = attn + row * 128;
  float v0 = p[lane], v1 = p[lane + 32], v2 = p[lane + 64], v3 = p[lane + 96];
  float mx = fmaxf(fmaxf(v0, v1), fmaxf(v2, v3));
#pragma unroll
  for (int off = 16; off; off >>= 1) mx = fmaxf(mx, __shfl_xor(mx, off, 32));
  const float e0 = __expf(v0 - mx), e1 = __expf(v1 - mx);
  const float e2 = __expf(v2 - mx), e3 = __expf(v3 - mx);
  float sum = e0 + e1 + e2 + e3;
#pragma unroll
  for (int off = 16; off; off >>= 1) sum += __shfl_xor(sum, off, 32);
  const float inv = 1.0f / sum;
  __bf16* o = abf + row * 128;
  o[lane]      = (__bf16)(e0 * inv);
  o[lane + 32] = (__bf16)(e1 * inv);
  o[lane + 64] = (__bf16)(e2 * inv);
  o[lane + 96] = (__bf16)(e3 * inv);
}

// ======================= criss-cross aggregation ============================
// out1[b,h,w,c] = sum_i aH[b,h,w,i] * v[b,i,w,c]   (block per (b,w) x c-tile)
__global__ __launch_bounds__(128) void agg_h_kernel(const __bf16* __restrict__ AT,
                                                    const __bf16* __restrict__ VT,
                                                    float* __restrict__ out1) {
  const int tid = threadIdx.x, lane = tid & 31, wid = tid >> 5;
  const int b = blockIdx.x >> 6, w = blockIdx.x & 63;
  const int nBase = blockIdx.y << 6;
  const int khalf = lane >> 4, kbase = khalf << 3;
  const int h = (wid << 4) + (lane & 15);

  const __bf16* arow = AT + (((size_t)((b << 6) + h) << 6) + w) * 128 + kbase;
  const __bf16* brow[4];
#pragma unroll
  for (int j = 0; j < 4; ++j) {
    const int c = nBase + (j << 4) + (lane & 15);
    brow[j] = VT + ((size_t)((b << 6) + w) * 512 + c) * 64 + (khalf << 3);
  }

  v8f acc[4];
  const v8f vz = {0.f, 0.f, 0.f, 0.f, 0.f, 0.f, 0.f, 0.f};
#pragma unroll
  for (int j = 0; j < 4; ++j) acc[j] = vz;
  gemm64_core(arow, brow, acc);

#pragma unroll
  for (int j = 0; j < 4; ++j) {
    const int c = nBase + (j << 4) + (lane & 15);
#pragma unroll
    for (int r = 0; r < 8; ++r) {
      const int hq = (wid << 4) + (khalf << 3) + r;
      out1[(((size_t)((b << 6) + hq) << 6) + w) * 512 + c] = acc[j][r];
    }
  }
}

// out2 + residual: feats_out = gamma*(out1 + sum_j aW*v) + feats_in
__global__ __launch_bounds__(128) void agg_w_kernel(const __bf16* __restrict__ AT,
                                                    const __bf16* __restrict__ VT,
                                                    const float* __restrict__ out1,
                                                    const __bf16* __restrict__ fin,
                                                    __bf16* __restrict__ fout,
                                                    const float* __restrict__ gamma) {
  const int tid = threadIdx.x, lane = tid & 31, wid = tid >> 5;
  const int b = blockIdx.x >> 6, h = blockIdx.x & 63;
  const int nBase = blockIdx.y << 6;
  const int khalf = lane >> 4, kbase = khalf << 3;
  const int w = (wid << 4) + (lane & 15);

  const __bf16* arow = AT + (((size_t)((b << 6) + h) << 6) + w) * 128 + 64 + kbase;
  const __bf16* brow[4];
#pragma unroll
  for (int j = 0; j < 4; ++j) {
    const int c = nBase + (j << 4) + (lane & 15);
    brow[j] = VT + ((size_t)((b << 6) + h) * 512 + c) * 64 + (khalf << 3);
  }

  v8f acc[4];
  const v8f vz = {0.f, 0.f, 0.f, 0.f, 0.f, 0.f, 0.f, 0.f};
#pragma unroll
  for (int j = 0; j < 4; ++j) acc[j] = vz;
  gemm64_core(arow, brow, acc);

  const float g = *gamma;
#pragma unroll
  for (int j = 0; j < 4; ++j) {
    const int c = nBase + (j << 4) + (lane & 15);
#pragma unroll
    for (int r = 0; r < 8; ++r) {
      const int wq = (wid << 4) + (khalf << 3) + r;
      const size_t mi = (((size_t)((b << 6) + h) << 6) + wq) * 512 + c;
      const float o = out1[mi] + acc[j][r];
      fout[mi] = (__bf16)(g * o + (float)fin[mi]);
    }
  }
}

// ================================ launcher ==================================

extern "C" void kernel_launch(void* const* d_in, const int* in_sizes, int n_in,
                              void* d_out, int out_size, void* d_ws, size_t ws_size,
                              hipStream_t stream) {
  const float* x     = (const float*)d_in[0];
  const float* w1    = (const float*)d_in[1];
  const float* bn1g  = (const float*)d_in[2];
  const float* bn1b  = (const float*)d_in[3];
  const float* bn1m  = (const float*)d_in[4];
  const float* bn1v  = (const float*)d_in[5];
  const float* qw    = (const float*)d_in[6];
  const float* qbias = (const float*)d_in[7];
  const float* kw    = (const float*)d_in[8];
  const float* kbias = (const float*)d_in[9];
  const float* vw    = (const float*)d_in[10];
  const float* vbias = (const float*)d_in[11];
  const float* gamma = (const float*)d_in[12];
  const float* w2    = (const float*)d_in[13];
  const float* bn2g  = (const float*)d_in[14];
  const float* bn2b  = (const float*)d_in[15];
  const float* bn2m  = (const float*)d_in[16];
  const float* bn2v  = (const float*)d_in[17];

  char* p = (char*)d_ws;
  auto alloc = [&](size_t bytes) -> char* {
    char* r = p;
    p += (bytes + 255) & ~(size_t)255;
    return r;
  };

  __bf16* xbf    = (__bf16*)alloc((size_t)16384 * 2048 * 2);   // x NHWC bf16
  __bf16* wt1    = (__bf16*)alloc((size_t)512 * 9 * 2048 * 2); // conv1 [N][9K]
  __bf16* wt2    = (__bf16*)alloc((size_t)512 * 9 * 512 * 2);  // conv2 [N][9K]
  __bf16* wqb    = (__bf16*)alloc((size_t)64 * 512 * 2);
  __bf16* wkb    = (__bf16*)alloc((size_t)64 * 512 * 2);
  __bf16* wvb    = (__bf16*)alloc((size_t)512 * 512 * 2);
  __bf16* featsA = (__bf16*)alloc((size_t)16384 * 512 * 2);
  __bf16* featsB = (__bf16*)alloc((size_t)16384 * 512 * 2);
  __bf16* qproj  = (__bf16*)alloc((size_t)16384 * 64 * 2);
  __bf16* kproj  = (__bf16*)alloc((size_t)16384 * 64 * 2);
  __bf16* vproj  = (__bf16*)alloc((size_t)16384 * 512 * 2);
  __bf16* vtH    = (__bf16*)alloc((size_t)16384 * 512 * 2);
  __bf16* vtW    = (__bf16*)alloc((size_t)16384 * 512 * 2);
  float*  attn   = (float*) alloc((size_t)16384 * 128 * 4);
  __bf16* abf    = (__bf16*)alloc((size_t)16384 * 128 * 2);
  float*  out1   = (float*) alloc((size_t)16384 * 512 * 4);

  const dim3 blk(128);

  // --- prep: bf16 conversions / weight reshapes ---
  cvt_x_kernel<<<131072, 256, 0, stream>>>(x, xbf);
  prep_w3_kernel<<<36864, 256, 0, stream>>>(w1, wt1, 2048, 512 * 2048 * 9);
  prep_w3_kernel<<<9216, 256, 0, stream>>>(w2, wt2, 512, 512 * 512 * 9);
  cvt_bf_kernel<<<128, 256, 0, stream>>>(qw, wqb, 64 * 512);
  cvt_bf_kernel<<<128, 256, 0, stream>>>(kw, wkb, 64 * 512);
  cvt_bf_kernel<<<1024, 256, 0, stream>>>(vw, wvb, 512 * 512);

  // --- conv1 (implicit GEMM, 9 taps) + BN + ReLU -> featsA ---
  gemm_nt_wmma<9, 2048, 1><<<dim3(128, 8), blk, 0, stream>>>(
      xbf, wt1, 512, nullptr, bn1g, bn1b, bn1m, bn1v, featsA, nullptr);

  // --- 2x criss-cross attention ---
  __bf16* fin = featsA;
  __bf16* fout = featsB;
  for (int it = 0; it < 2; ++it) {
    gemm_nt_wmma<1, 512, 0><<<dim3(128, 1), blk, 0, stream>>>(
        fin, wqb, 64, qbias, nullptr, nullptr, nullptr, nullptr,
        qproj, nullptr);
    gemm_nt_wmma<1, 512, 0><<<dim3(128, 1), blk, 0, stream>>>(
        fin, wkb, 64, kbias, nullptr, nullptr, nullptr, nullptr,
        kproj, nullptr);
    gemm_nt_wmma<1, 512, 0><<<dim3(128, 8), blk, 0, stream>>>(
        fin, wvb, 512, vbias, nullptr, nullptr, nullptr, nullptr,
        vproj, nullptr);
    transpose_v_kernel<<<32768, 256, 0, stream>>>(vproj, vtH, vtW);
    energy_h_kernel<<<256, blk, 0, stream>>>(qproj, kproj, attn);
    energy_w_kernel<<<256, blk, 0, stream>>>(qproj, kproj, attn);
    softmax_kernel<<<4096, blk, 0, stream>>>(attn, abf);
    agg_h_kernel<<<dim3(256, 8), blk, 0, stream>>>(abf, vtH, out1);
    agg_w_kernel<<<dim3(256, 8), blk, 0, stream>>>(abf, vtW, out1, fin, fout,
                                                   gamma);
    __bf16* t = fin; fin = fout; fout = t;
  }

  // --- conv2 (implicit GEMM) + BN + ReLU -> d_out (NCHW f32) ---
  gemm_nt_wmma<9, 512, 2><<<dim3(128, 8), blk, 0, stream>>>(
      fin, wt2, 512, nullptr, bn2g, bn2b, bn2m, bn2v, nullptr, (float*)d_out);

  (void)in_sizes; (void)n_in; (void)out_size; (void)ws_size;
}
